// CABASC_54958401519957
// MI455X (gfx1250) — matrix operand order repaired and checked by reading, not compiled, not run on hardware
//
#include <hip/hip_runtime.h>
#include <hip/hip_bf16.h>
#include <cstdint>

#define Bn 128
#define Ln 512
#define LAn 8
#define En 300
#define Hn 300
#define G3n 900
#define GNn 912     // padded 3H (mult of 16)
#define KPn 320     // padded E/H for WMMA K (mult of 32)
#define NKX 304     // padded E for kx N (mult of 16)
#define Pn 3

typedef __attribute__((ext_vector_type(16))) _Float16 v16h;
typedef __attribute__((ext_vector_type(8)))  float    v8f;
typedef __attribute__((ext_vector_type(4)))  unsigned int v4u;
typedef __attribute__((ext_vector_type(8)))  int      v8i;
typedef __attribute__((ext_vector_type(4)))  int      v4i;

union FragU { v16h h; uint32_t u[8]; };

#if defined(__gfx1250__) && __has_builtin(__builtin_amdgcn_tensor_load_to_lds)
#define HAVE_TDM 1
#else
#define HAVE_TDM 0
#endif

// ---- WMMA fragment loaders (wave32, 16x16x32 f16; contiguous K order k = 16*(lane/16) + s in A and B) ----

// A fragment: A row-major [M x K], 16x32 tile at (m0, k0), row stride lda (elements).
__device__ __forceinline__ v16h load_afrag(const _Float16* A, long lda, int m0, int k0) {
  int lane = threadIdx.x & 31;
  int m  = lane & 15;
  int kb = (lane & 16) ? 16 : 0;        // lanes 0-15: K=0..15, lanes 16-31: K=16..31
  const _Float16* r = A + (long)(m0 + m) * lda + k0;
  FragU f;
#pragma unroll
  for (int i = 0; i < 8; ++i) {
    int k = kb + 2 * i;
    f.u[i] = *(const uint32_t*)(r + k);
  }
  return f.h;
}

// B fragment for C = A * W^T: logical B[k][n] = W[n0+n][k0+k], W row-major [N x K].
__device__ __forceinline__ v16h load_bfrag_wt(const _Float16* W, long ldw, int n0, int k0) {
  int lane = threadIdx.x & 31;
  int n  = lane & 15;
  int kb = (lane & 16) ? 16 : 0;        // lanes 0-15: K=0..15, lanes 16-31: K=16..31
  const _Float16* r = W + (long)(n0 + n) * ldw + k0 + kb;
  FragU f;
#pragma unroll
  for (int i = 0; i < 8; ++i) f.u[i] = *(const uint32_t*)(r + 2 * i);
  return f.h;
}

__device__ __forceinline__ float sigmoidf_(float x) { return 1.0f / (1.0f + expf(-x)); }

#if HAVE_TDM
// Issue a TDM 2-D tile load: rows x tile_k (f16 elements) from row-major gsrc
// (row stride = tensor_d0 elements) into LDS at byte offset lds_off.
// D# layout per CDNA5 ISA ch.8 (group0: count/lds/global/type; group1: dims/tiles/strides).
__device__ __forceinline__ void tdm_load_tile(const _Float16* gsrc, uint32_t lds_off,
                                              int rows, int tensor_d0, int tile_k) {
  unsigned long long ga = (unsigned long long)(uintptr_t)gsrc;
  v4u g0;
  g0.x = 1u;                                              // count=1, user descriptor
  g0.y = lds_off;                                         // lds_addr [63:32]
  g0.z = (uint32_t)ga;                                    // global_addr [95:64]
  g0.w = (uint32_t)((ga >> 32) & 0x1FFFFFFu) | (2u << 30); // addr[56:32] | type=2
  v8i g1;
  g1[0] = (1 << 16);                                      // data_size=1 (2 bytes)
  g1[1] = (tensor_d0 & 0xFFFF) << 16;                     // tensor_dim0[15:0] @ bits 63:48
  g1[2] = ((tensor_d0 >> 16) & 0xFFFF) | ((rows & 0xFFFF) << 16);  // dim0 hi | dim1 lo
  g1[3] = ((rows >> 16) & 0xFFFF) | ((tile_k & 0xFFFF) << 16);     // dim1 hi | tile_dim0
  g1[4] = (rows & 0xFFFF);                                // tile_dim1 (tile_dim2 = 0)
  g1[5] = tensor_d0;                                      // tensor_dim0_stride[31:0]
  g1[6] = 0;                                              // stride hi | dim1_stride lo
  g1[7] = 0;
  v4i gz = {0, 0, 0, 0};
#if __clang_major__ >= 23
  v8i gz8 = {0, 0, 0, 0, 0, 0, 0, 0};
  __builtin_amdgcn_tensor_load_to_lds(g0, g1, gz, gz, gz8, 0);
#else
  __builtin_amdgcn_tensor_load_to_lds(g0, g1, gz, gz, 0);
#endif
}
#endif

// ---------------- K0: sequence lengths ----------------
__global__ void lengths_kernel(const int* __restrict__ text, const int* __restrict__ asp,
                               const int* __restrict__ xl, const int* __restrict__ xr,
                               int* mem_len, int* asp_len, int* left_len, int* right_len) {
  int b = threadIdx.x;
  if (b >= Bn) return;
  int c = 0;
  for (int t = 0; t < Ln; ++t) c += (text[b * Ln + t] != 0);
  mem_len[b] = c;
  c = 0;
  for (int i = 0; i < LAn; ++i) c += (asp[b * LAn + i] != 0);
  asp_len[b] = c;
  c = 0;
  for (int t = 0; t < Ln; ++t) c += (xl[b * Ln + t] != 0);
  left_len[b] = c;
  c = 0;
  for (int t = 0; t < Ln; ++t) c += (xr[b * Ln + t] != 0);
  right_len[b] = c;
}

// ---------------- pack f32 weights into zero-padded f16 ----------------
__global__ void pack_f16(const float* __restrict__ src, _Float16* __restrict__ dst,
                         int rows, int cols, int dcols, long n) {
  long i = (long)blockIdx.x * blockDim.x + threadIdx.x;
  if (i >= n) return;
  int r = (int)(i / dcols), c = (int)(i % dcols);
  dst[i] = (r < rows && c < cols) ? (_Float16)src[(long)r * cols + c] : (_Float16)0.0f;
}

__global__ void zero_f32(float* p, long n) {
  long i = (long)blockIdx.x * blockDim.x + threadIdx.x;
  if (i < n) p[i] = 0.0f;
}
__global__ void zero_f16(_Float16* p, long n) {
  long i = (long)blockIdx.x * blockDim.x + threadIdx.x;
  if (i < n) p[i] = (_Float16)0.0f;
}

// ---------------- K1: embedding gathers for both GRU inputs (f32 -> f16, K-padded) ----------------
__global__ void gather_gru_inputs(const int* __restrict__ xl, const int* __restrict__ xr,
                                  const float* __restrict__ emb,
                                  _Float16* __restrict__ xl16, _Float16* __restrict__ xr16) {
  int row = blockIdx.x;           // b*L + t
  int e = threadIdx.x;            // 0..319
  long o = (long)row * KPn + e;
  int idl = xl[row], idr = xr[row];
  const float* rl = emb + (long)idl * En;
  const float* rr = emb + (long)idr * En;
  __builtin_prefetch(rr + e, 0, 0);   // global_prefetch_b8
  xl16[o] = (e < En) ? (_Float16)rl[e] : (_Float16)0.0f;
  xr16[o] = (e < En) ? (_Float16)rr[e] : (_Float16)0.0f;
}

// ---------------- K2: aspect mean, qx = aspect@w_q^T+b_q, qdot = qx . attn_w[E:] ----------------
__global__ void aspect_qdot(const int* __restrict__ asp_idx, const float* __restrict__ emb_aspect,
                            const int* __restrict__ asp_len, const float* __restrict__ w_q,
                            const float* __restrict__ b_q, const float* __restrict__ attn_w,
                            float* __restrict__ qdot) {
  __shared__ float av[En];
  __shared__ float red[512];
  int b = blockIdx.x, tid = threadIdx.x;
  int al = asp_len[b];
  if (tid < En) {
    float s = 0.0f;
    for (int i = 0; i < LAn; ++i)
      if (i < al) s += emb_aspect[(long)asp_idx[b * LAn + i] * En + tid];
    av[tid] = s / (float)al;
  }
  __syncthreads();
  float contrib = 0.0f;
  if (tid < En) {
    float q = b_q[tid];
    for (int e = 0; e < En; ++e) q += av[e] * w_q[tid * En + e];
    contrib = q * attn_w[En + tid];
  }
  red[tid] = contrib;
  if (tid < 192) red[320 + tid] = 0.0f;
  __syncthreads();
  for (int s = 256; s > 0; s >>= 1) {
    if (tid < s) red[tid] += red[tid + s];
    __syncthreads();
  }
  if (tid == 0) qdot[b] = red[0];
}

// ---------------- K3: per-timestep WMMA GEMM (gi/gh for both GRUs) ----------------
// 4 mats x (8 M-tiles x 57 N-tiles) = 1824 wave-tiles; 228 blocks x 8 waves.
__global__ void gru_step_gemm(const _Float16* __restrict__ xl16, const _Float16* __restrict__ xr16,
                              const _Float16* __restrict__ hl16, const _Float16* __restrict__ hr16,
                              const _Float16* __restrict__ Wih_l, const _Float16* __restrict__ Whh_l,
                              const _Float16* __restrict__ Wih_r, const _Float16* __restrict__ Whh_r,
                              float* __restrict__ gi_l, float* __restrict__ gh_l,
                              float* __restrict__ gi_r, float* __restrict__ gh_r, int t) {
  int wave = (int)((blockIdx.x * blockDim.x + threadIdx.x) >> 5);  // 0..1823
  int mat = wave / 456;
  int rem = wave % 456;
  int mt = rem / 57, nt = rem % 57;
  int m0 = mt * 16, n0 = nt * 16;

  const _Float16* A;
  long lda;
  const _Float16* Wm;
  float* Out;
  if (mat == 0)      { A = xl16 + (long)t * KPn; lda = (long)Ln * KPn; Wm = Wih_l; Out = gi_l; }
  else if (mat == 1) { A = hl16;                 lda = KPn;            Wm = Whh_l; Out = gh_l; }
  else if (mat == 2) { A = xr16 + (long)t * KPn; lda = (long)Ln * KPn; Wm = Wih_r; Out = gi_r; }
  else               { A = hr16;                 lda = KPn;            Wm = Whh_r; Out = gh_r; }

  v8f c = {};
#pragma unroll
  for (int kk = 0; kk < KPn; kk += 32) {
    v16h a = load_afrag(A, lda, m0, kk);
    v16h b = load_bfrag_wt(Wm, KPn, n0, kk);
    c = __builtin_amdgcn_wmma_f32_16x16x32_f16(false, a, false, b, (short)0, c, false, false);
  }
  int lane = threadIdx.x & 31;
  int n = lane & 15;
  int mo = (lane & 16) ? 8 : 0;
#pragma unroll
  for (int i = 0; i < 8; ++i)
    Out[(long)(m0 + mo + i) * GNn + n0 + n] = c[i];
}

// ---------------- K4: per-timestep gate nonlinearity + attn dot ----------------
__global__ void gru_step_gate(const float* __restrict__ gi_l, const float* __restrict__ gh_l,
                              const float* __restrict__ gi_r, const float* __restrict__ gh_r,
                              const float* __restrict__ b_ih_l, const float* __restrict__ b_hh_l,
                              const float* __restrict__ b_ih_r, const float* __restrict__ b_hh_r,
                              float* __restrict__ hl32, float* __restrict__ hr32,
                              _Float16* __restrict__ hl16, _Float16* __restrict__ hr16,
                              const float* __restrict__ mlp_l_w, const float* __restrict__ mlp_l_b,
                              const float* __restrict__ mlp_r_w, const float* __restrict__ mlp_r_b,
                              const int* __restrict__ left_len, const int* __restrict__ right_len,
                              float* __restrict__ attn_l, float* __restrict__ attn_r, int t) {
  __shared__ float redl[512];
  __shared__ float redr[512];
  int b = blockIdx.x, j = threadIdx.x;
  float cl = 0.0f, cr = 0.0f;
  if (j < Hn) {
    {
      float gir = gi_l[b * GNn + j]          + b_ih_l[j];
      float giz = gi_l[b * GNn + Hn + j]     + b_ih_l[Hn + j];
      float gin = gi_l[b * GNn + 2 * Hn + j] + b_ih_l[2 * Hn + j];
      float ghr = gh_l[b * GNn + j]          + b_hh_l[j];
      float ghz = gh_l[b * GNn + Hn + j]     + b_hh_l[Hn + j];
      float ghn = gh_l[b * GNn + 2 * Hn + j] + b_hh_l[2 * Hn + j];
      float r = sigmoidf_(gir + ghr);
      float z = sigmoidf_(giz + ghz);
      float nn = tanhf(gin + r * ghn);
      float hp = hl32[b * Hn + j];
      float h = (1.0f - z) * nn + z * hp;
      hl32[b * Hn + j] = h;
      hl16[b * KPn + j] = (_Float16)h;
      cl = h * mlp_l_w[j];
    }
    {
      float gir = gi_r[b * GNn + j]          + b_ih_r[j];
      float giz = gi_r[b * GNn + Hn + j]     + b_ih_r[Hn + j];
      float gin = gi_r[b * GNn + 2 * Hn + j] + b_ih_r[2 * Hn + j];
      float ghr = gh_r[b * GNn + j]          + b_hh_r[j];
      float ghz = gh_r[b * GNn + Hn + j]     + b_hh_r[Hn + j];
      float ghn = gh_r[b * GNn + 2 * Hn + j] + b_hh_r[2 * Hn + j];
      float r = sigmoidf_(gir + ghr);
      float z = sigmoidf_(giz + ghz);
      float nn = tanhf(gin + r * ghn);
      float hp = hr32[b * Hn + j];
      float h = (1.0f - z) * nn + z * hp;
      hr32[b * Hn + j] = h;
      hr16[b * KPn + j] = (_Float16)h;
      cr = h * mlp_r_w[j];
    }
  } else {
    hl16[b * KPn + j] = (_Float16)0.0f;   // keep K padding zero
    hr16[b * KPn + j] = (_Float16)0.0f;
  }
  redl[j] = cl; redr[j] = cr;
  if (j < 192) { redl[320 + j] = 0.0f; redr[320 + j] = 0.0f; }
  __syncthreads();
  for (int s = 256; s > 0; s >>= 1) {
    if (j < s) { redl[j] += redl[j + s]; redr[j] += redr[j + s]; }
    __syncthreads();
  }
  if (j == 0) {
    float dl = (t < left_len[b]) ? redl[0] : 0.0f;   // padded ctx outputs are zeroed in ref
    attn_l[b * Ln + t] = sigmoidf_(dl + mlp_l_b[0]) + 0.5f;
    float dr = (t < right_len[b]) ? redr[0] : 0.0f;
    attn_r[b * Ln + t] = sigmoidf_(dr + mlp_r_b[0]) + 0.5f;
  }
}

// ---------------- K5: position-wise scale, write scaled memory in f16 (K-padded) ----------------
__global__ void memory_scale(const int* __restrict__ text, const float* __restrict__ emb,
                             const int* __restrict__ mem_len, const int* __restrict__ asp_len,
                             const int* __restrict__ left_len,
                             const float* __restrict__ attn_l, const float* __restrict__ attn_r,
                             float* __restrict__ scale_out, _Float16* __restrict__ mem16) {
  int row = blockIdx.x;               // b*L + t
  int b = row >> 9, t = row & 511;
  int e = threadIdx.x;
  int ml = mem_len[b];
  float s;
  if (t >= ml) {
    s = 1.0f;
  } else {
    int a_start = left_len[b] - asp_len[b];
    int a_end = left_len[b];
    int ri = t - a_start;
    ri = ri < 0 ? 0 : (ri > Ln - 1 ? Ln - 1 : ri);
    float ar = attn_r[b * Ln + ri];
    if (t < a_start)      s = attn_l[b * Ln + t];
    else if (t < a_end)   s = attn_l[b * Ln + t] + ar;
    else                  s = ar;
  }
  if (e == 0) scale_out[row] = s;
  float v = 0.0f;
  if (e < En && t < ml) v = emb[(long)text[row] * En + e] * s;
  mem16[(long)row * KPn + e] = (_Float16)v;
}

// ---------------- K6: v_s = sum_t(scaled memory)/mem_len (f32 path for accuracy) ----------------
__global__ void vs_kernel(const int* __restrict__ text, const float* __restrict__ emb,
                          const int* __restrict__ mem_len, const float* __restrict__ scale,
                          float* __restrict__ v_s) {
  int b = blockIdx.x, e = threadIdx.x;
  if (e >= En) return;
  int ml = mem_len[b];
  float acc = 0.0f;
  for (int t = 0; t < ml; ++t)
    acc += emb[(long)text[b * Ln + t] * En + e] * scale[b * Ln + t];
  v_s[b * En + e] = acc / (float)ml;
}

// ---------------- K7: kx = memory @ w_k^T + b_k via TDM-staged LDS WMMA ----------------
// Block = 8 waves, one 128-row band. Per K-chunk: TDM stages W slice (304 x 32 f16)
// into LDS once; each wave loads its A fragment once and reuses it for all 19 N-tiles
// (19 f32 accumulators = 152 VGPRs, fine under wave32's 1024-VGPR budget).
__global__ void kx_gemm_lds(const _Float16* __restrict__ mem16, const _Float16* __restrict__ wk16,
                            const float* __restrict__ b_k, float* __restrict__ kx) {
  __shared__ _Float16 Wsh[NKX * 32];          // 304 rows x 32 K-halves = 19456 B
  int wv = threadIdx.x >> 5;
  int lane = threadIdx.x & 31;
  int m0 = (blockIdx.x * 8 + wv) * 16;

  v8f acc[19];
#pragma unroll
  for (int nt = 0; nt < 19; ++nt)
#pragma unroll
    for (int i = 0; i < 8; ++i) acc[nt][i] = 0.0f;

  for (int kk = 0; kk < KPn; kk += 32) {
    __syncthreads();                          // previous iteration's LDS reads done
#if HAVE_TDM
    if (wv == 0) {
      tdm_load_tile(wk16 + kk, (uint32_t)(uintptr_t)(void*)Wsh, NKX, KPn, 32);
#if __has_builtin(__builtin_amdgcn_s_wait_tensorcnt)
      __builtin_amdgcn_s_wait_tensorcnt(0);
#endif
    }
#else
    // Fallback: cooperative 16B-chunk copy (1216 chunks over 256 threads).
    for (int c = threadIdx.x; c < (NKX * 32) / 8; c += 256) {
      int row = c >> 2, cc = c & 3;
      ((int4*)Wsh)[c] = *(const int4*)(wk16 + (long)row * KPn + kk + cc * 8);
    }
#endif
    __syncthreads();

    v16h a = load_afrag(mem16, KPn, m0, kk);
    int n = lane & 15;
    int kb = (lane & 16) ? 16 : 0;
#pragma unroll
    for (int nt = 0; nt < 19; ++nt) {
      const _Float16* r = &Wsh[(nt * 16 + n) * 32 + kb];
      FragU f;
#pragma unroll
      for (int i = 0; i < 8; ++i) f.u[i] = *(const uint32_t*)(r + 2 * i);
      acc[nt] = __builtin_amdgcn_wmma_f32_16x16x32_f16(false, a, false, f.h,
                                                       (short)0, acc[nt], false, false);
    }
  }

  int n = lane & 15;
  int mo = (lane & 16) ? 8 : 0;
#pragma unroll
  for (int nt = 0; nt < 19; ++nt) {
    int gn = nt * 16 + n;
    float bias = (gn < En) ? b_k[gn] : 0.0f;
#pragma unroll
    for (int i = 0; i < 8; ++i)
      kx[(long)(m0 + mo + i) * NKX + gn] = acc[nt][i] + bias;
  }
}

// ---------------- K8: score = tanh(kx . attn_w[:E] + qdot[b]) — one wave per row ----------------
__global__ void score_kernel(const float* __restrict__ kx, const float* __restrict__ attn_w,
                             const float* __restrict__ qdot, float* __restrict__ score) {
  int wave = (int)((blockIdx.x * blockDim.x + threadIdx.x) >> 5);  // 0..65535 = b*L+t
  int lane = threadIdx.x & 31;
  float acc = 0.0f;
#pragma unroll
  for (int i = 0; i < 10; ++i) {
    int col = lane + 32 * i;
    if (col < En) acc += kx[(long)wave * NKX + col] * attn_w[col];
  }
  for (int m = 16; m >= 1; m >>= 1) acc += __shfl_xor(acc, m, 32);
  if (lane == 0) score[wave] = tanhf(acc + qdot[wave >> 9]);
}

// ---------------- K9: softmax over L per batch ----------------
__global__ void softmax_kernel(const float* __restrict__ score, float* __restrict__ probs) {
  __shared__ float red[512];
  int b = blockIdx.x, t = threadIdx.x;
  float s = score[b * Ln + t];
  red[t] = s;
  __syncthreads();
  for (int k = 256; k > 0; k >>= 1) {
    if (t < k) red[t] = fmaxf(red[t], red[t + k]);
    __syncthreads();
  }
  float mx = red[0];
  __syncthreads();
  float e = expf(s - mx);
  red[t] = e;
  __syncthreads();
  for (int k = 256; k > 0; k >>= 1) {
    if (t < k) red[t] += red[t + k];
    __syncthreads();
  }
  probs[b * Ln + t] = e / red[0];
}

// ---------------- K10: v_ts -> proj -> +v_s -> mlp tanh -> dense (fused per batch) ----------------
__global__ void final_kernel(const float* __restrict__ probs, const float* __restrict__ kx,
                             const float* __restrict__ v_s,
                             const float* __restrict__ proj_w, const float* __restrict__ proj_b,
                             const float* __restrict__ mlp_w, const float* __restrict__ mlp_b,
                             const float* __restrict__ dense_w, const float* __restrict__ dense_b,
                             float* __restrict__ out) {
  __shared__ float pre[En];
  __shared__ float vns[En];
  __shared__ float vms[En];
  int b = blockIdx.x, tid = threadIdx.x;
  if (tid < En) {
    float acc = 0.0f;
    for (int t = 0; t < Ln; ++t)
      acc += probs[b * Ln + t] * kx[((long)b * Ln + t) * NKX + tid];
    pre[tid] = acc;
  }
  __syncthreads();
  if (tid < En) {
    float v = proj_b[tid];
    for (int e = 0; e < En; ++e) v += proj_w[tid * En + e] * pre[e];
    vns[tid] = v + v_s[b * En + tid];
  }
  __syncthreads();
  if (tid < En) {
    float v = mlp_b[tid];
    for (int e = 0; e < En; ++e) v += mlp_w[tid * En + e] * vns[e];
    vms[tid] = tanhf(v);
  }
  __syncthreads();
  if (tid < Pn) {
    float v = dense_b[tid];
    for (int e = 0; e < En; ++e) v += dense_w[tid * En + e] * vms[e];
    out[b * Pn + tid] = v;
  }
}

// =====================================================================================

extern "C" void kernel_launch(void* const* d_in, const int* in_sizes, int n_in,
                              void* d_out, int out_size, void* d_ws, size_t ws_size,
                              hipStream_t stream) {
  (void)in_sizes; (void)n_in; (void)out_size; (void)ws_size;
  const int*   text       = (const int*)d_in[0];
  const int*   aspi       = (const int*)d_in[1];
  const int*   xl         = (const int*)d_in[2];
  const int*   xr         = (const int*)d_in[3];
  const float* emb        = (const float*)d_in[4];
  const float* emb_aspect = (const float*)d_in[5];
  const float* W_ih_l     = (const float*)d_in[6];
  const float* W_hh_l     = (const float*)d_in[7];
  const float* b_ih_l     = (const float*)d_in[8];
  const float* b_hh_l     = (const float*)d_in[9];
  const float* W_ih_r     = (const float*)d_in[10];
  const float* W_hh_r     = (const float*)d_in[11];
  const float* b_ih_r     = (const float*)d_in[12];
  const float* b_hh_r     = (const float*)d_in[13];
  const float* mlp_l_w    = (const float*)d_in[14];
  const float* mlp_l_b    = (const float*)d_in[15];
  const float* mlp_r_w    = (const float*)d_in[16];
  const float* mlp_r_b    = (const float*)d_in[17];
  const float* w_k        = (const float*)d_in[18];
  const float* b_k        = (const float*)d_in[19];
  const float* w_q        = (const float*)d_in[20];
  const float* b_q        = (const float*)d_in[21];
  const float* attn_w     = (const float*)d_in[22];
  const float* proj_w     = (const float*)d_in[23];
  const float* proj_b     = (const float*)d_in[24];
  const float* mlp_w      = (const float*)d_in[25];
  const float* mlp_b      = (const float*)d_in[26];
  const float* dense_w    = (const float*)d_in[27];
  const float* dense_b    = (const float*)d_in[28];
  float* out = (float*)d_out;

  char* ws = (char*)d_ws;
  size_t off = 0;
  auto alloc = [&](size_t bytes) -> void* {
    void* p = ws + off;
    off = (off + bytes + 255) & ~(size_t)255;
    return p;
  };

  int* mem_len   = (int*)alloc(Bn * 4);
  int* asp_len   = (int*)alloc(Bn * 4);
  int* left_len  = (int*)alloc(Bn * 4);
  int* right_len = (int*)alloc(Bn * 4);
  float* qdot    = (float*)alloc(Bn * 4);
  float* attn_l  = (float*)alloc((size_t)Bn * Ln * 4);
  float* attn_r  = (float*)alloc((size_t)Bn * Ln * 4);
  float* scale   = (float*)alloc((size_t)Bn * Ln * 4);
  float* score   = (float*)alloc((size_t)Bn * Ln * 4);
  float* probs   = (float*)alloc((size_t)Bn * Ln * 4);
  float* hl32    = (float*)alloc((size_t)Bn * Hn * 4);
  float* hr32    = (float*)alloc((size_t)Bn * Hn * 4);
  _Float16* hl16 = (_Float16*)alloc((size_t)Bn * KPn * 2);
  _Float16* hr16 = (_Float16*)alloc((size_t)Bn * KPn * 2);
  float* gi_l    = (float*)alloc((size_t)Bn * GNn * 4);
  float* gh_l    = (float*)alloc((size_t)Bn * GNn * 4);
  float* gi_r    = (float*)alloc((size_t)Bn * GNn * 4);
  float* gh_r    = (float*)alloc((size_t)Bn * GNn * 4);
  _Float16* Wih_l16 = (_Float16*)alloc((size_t)GNn * KPn * 2);
  _Float16* Whh_l16 = (_Float16*)alloc((size_t)GNn * KPn * 2);
  _Float16* Wih_r16 = (_Float16*)alloc((size_t)GNn * KPn * 2);
  _Float16* Whh_r16 = (_Float16*)alloc((size_t)GNn * KPn * 2);
  _Float16* wk16    = (_Float16*)alloc((size_t)NKX * KPn * 2);
  _Float16* xl16    = (_Float16*)alloc((size_t)Bn * Ln * KPn * 2);
  _Float16* xr16    = (_Float16*)alloc((size_t)Bn * Ln * KPn * 2);
  _Float16* mem16   = (_Float16*)alloc((size_t)Bn * Ln * KPn * 2);
  float* kx         = (float*)alloc((size_t)Bn * Ln * NKX * 4);
  float* v_s        = (float*)alloc((size_t)Bn * En * 4);

  // K0: lengths
  lengths_kernel<<<1, 128, 0, stream>>>(text, aspi, xl, xr, mem_len, asp_len, left_len, right_len);

  // pack weights to padded f16
  {
    long n = (long)GNn * KPn;
    int blocks = (int)((n + 255) / 256);
    pack_f16<<<blocks, 256, 0, stream>>>(W_ih_l, Wih_l16, G3n, En, KPn, n);
    pack_f16<<<blocks, 256, 0, stream>>>(W_hh_l, Whh_l16, G3n, Hn, KPn, n);
    pack_f16<<<blocks, 256, 0, stream>>>(W_ih_r, Wih_r16, G3n, En, KPn, n);
    pack_f16<<<blocks, 256, 0, stream>>>(W_hh_r, Whh_r16, G3n, Hn, KPn, n);
    long nk = (long)NKX * KPn;
    pack_f16<<<(int)((nk + 255) / 256), 256, 0, stream>>>(w_k, wk16, En, En, KPn, nk);
  }

  // init h state
  zero_f32<<<(Bn * Hn + 255) / 256, 256, 0, stream>>>(hl32, (long)Bn * Hn);
  zero_f32<<<(Bn * Hn + 255) / 256, 256, 0, stream>>>(hr32, (long)Bn * Hn);
  zero_f16<<<(Bn * KPn + 255) / 256, 256, 0, stream>>>(hl16, (long)Bn * KPn);
  zero_f16<<<(Bn * KPn + 255) / 256, 256, 0, stream>>>(hr16, (long)Bn * KPn);

  // K1: embedding gathers
  gather_gru_inputs<<<Bn * Ln, KPn, 0, stream>>>(xl, xr, emb, xl16, xr16);

  // K2: aspect + qdot
  aspect_qdot<<<Bn, KPn, 0, stream>>>(aspi, emb_aspect, asp_len, w_q, b_q, attn_w, qdot);

  // GRU time loop: 512 steps x (WMMA GEMM + fused gates/attn)
  for (int t = 0; t < Ln; ++t) {
    gru_step_gemm<<<228, 256, 0, stream>>>(xl16, xr16, hl16, hr16,
                                           Wih_l16, Whh_l16, Wih_r16, Whh_r16,
                                           gi_l, gh_l, gi_r, gh_r, t);
    gru_step_gate<<<Bn, KPn, 0, stream>>>(gi_l, gh_l, gi_r, gh_r,
                                          b_ih_l, b_hh_l, b_ih_r, b_hh_r,
                                          hl32, hr32, hl16, hr16,
                                          mlp_l_w, mlp_l_b, mlp_r_w, mlp_r_b,
                                          left_len, right_len, attn_l, attn_r, t);
  }

  // K5: scale + scaled memory (f16)
  memory_scale<<<Bn * Ln, KPn, 0, stream>>>(text, emb, mem_len, asp_len, left_len,
                                            attn_l, attn_r, scale, mem16);
  // K6: v_s
  vs_kernel<<<Bn, KPn, 0, stream>>>(text, emb, mem_len, scale, v_s);

  // K7: kx GEMM (TDM-staged LDS + WMMA), 512 blocks x 8 waves
  kx_gemm_lds<<<512, 256, 0, stream>>>(mem16, wk16, b_k, kx);

  // K8: score
  score_kernel<<<8192, 256, 0, stream>>>(kx, attn_w, qdot, score);

  // K9: softmax
  softmax_kernel<<<Bn, 512, 0, stream>>>(score, probs);

  // K10: fused tail -> logits
  final_kernel<<<Bn, KPn, 0, stream>>>(probs, kx, v_s, proj_w, proj_b,
                                       mlp_w, mlp_b, dense_w, dense_b, out);
}
